// SelfAttention_5626407157938
// MI455X (gfx1250) — matrix operand order repaired and checked by reading, not compiled, not run on hardware
//
#include <hip/hip_runtime.h>

// ---------------------------------------------------------------------------
// SAGAN self-attention for MI455X (gfx1250, wave32, WMMA bf16 + TDM).
// Pipeline: cvt x->bf16, pack weights->bf16, proj GEMM (f/g/h), logits GEMM,
// row softmax, output GEMM (double-buffered A-tiles via tensor_load_to_lds,
// TENSORcnt split-phase wait) with fused gamma*sa + x epilogue.
// ---------------------------------------------------------------------------

typedef __attribute__((ext_vector_type(16))) __bf16 bf16x16;
typedef __attribute__((ext_vector_type(8)))  __bf16 bf16x8;
typedef __attribute__((ext_vector_type(2)))  __bf16 bf16x2;
typedef __attribute__((ext_vector_type(8)))  float  f32x8;
typedef __attribute__((ext_vector_type(4)))  float  f32x4;
typedef __attribute__((ext_vector_type(4)))  unsigned int u32x4;
typedef __attribute__((ext_vector_type(8)))  unsigned int u32x8;

#define LDK 40  // LDS row stride for 32-wide K tiles (32 + 8 pad halves)

// Problem constants
#define NB   16
#define CIN  512
#define WH   9216
#define MCAT 640      // 64 (f) + 64 (g) + 512 (h) output channels
#define LSEQ 1536
#define KF   384
#define KH   3072

// --- WMMA fragment loaders (ISA 7.12.2 layouts, wave32) ---------------------
static __device__ __forceinline__ bf16x16 frag_a(const __bf16* tile, int row0, int lane) {
    const int lr = lane & 15, hi = lane >> 4;
    const __bf16* p = tile + (row0 + lr) * LDK + hi * 8;
    bf16x8 lo = *(const bf16x8*)(p);
    bf16x8 hv = *(const bf16x8*)(p + 16);
    return __builtin_shufflevector(lo, hv, 0,1,2,3,4,5,6,7,8,9,10,11,12,13,14,15);
}
static __device__ __forceinline__ bf16x16 frag_b(const __bf16* tile, int row0, int lane) {
    const int lr = lane & 15, hi = lane >> 4;
    const __bf16* p = tile + (row0 + lr) * LDK + hi * 8;
    bf16x8 lo = *(const bf16x8*)(p);
    bf16x8 hv = *(const bf16x8*)(p + 16);
    return __builtin_shufflevector(lo, hv, 0,1,2,3,4,5,6,7,8,9,10,11,12,13,14,15);
}
static __device__ __forceinline__ f32x8 mma_bf16(bf16x16 a, bf16x16 b, f32x8 c) {
    return __builtin_amdgcn_wmma_f32_16x16x32_bf16(false, a, false, b, (short)0, c,
                                                   false, false);
}

// --- TDM: async DMA of a 2D bf16 tile (32 x 128 rows) into padded LDS ------
// D# per ISA ch.8: group0 = {count/flags, lds_addr, global_addr, type=2};
// group1 packs data_size=2B, pad (16 DWORDs interval, +4 DWORDs) so the LDS
// row stride becomes 40 halves == LDK, tensor dims/strides, tile dims.
static __device__ __forceinline__ void tdm_load_tile_a(unsigned lds_off,
                                                       unsigned long long gaddr) {
    u32x4 g0;
    g0[0] = 1u;                                       // count=1, user descriptor
    g0[1] = lds_off;                                  // LDS byte address
    g0[2] = (unsigned)gaddr;                          // global_addr[31:0]
    g0[3] = ((unsigned)(gaddr >> 32) & 0x1FFFFFFu) | (2u << 30); // [56:32] | type=2
    u32x8 g1;
    g1[0] = (1u << 16)        // data_size = 1 -> 2 bytes
          | (1u << 20)        // pad_enable
          | (3u << 22)        // pad_interval: 16 DWORDs (one 32-half row)
          | (3u << 25);       // pad_amount: 4 DWORDs (8 halves) -> stride 40
    g1[1] = (unsigned)LSEQ << 16;            // tensor_dim0 = 1536 (low16 @ bit48)
    g1[2] = (unsigned)KH << 16;              // tensor_dim1 = 3072 (low16 @ bit80)
    g1[3] = 32u << 16;                       // tile_dim0 = 32 elements
    g1[4] = 128u;                            // tile_dim1 = 128 rows, tile_dim2 = 0
    g1[5] = (unsigned)LSEQ;                  // tensor_dim0_stride = 1536 elements
    g1[6] = 0u;
    g1[7] = 0u;
    asm volatile("tensor_load_to_lds %0, %1" :: "s"(g0), "s"(g1) : "memory");
}

// --- K0: x (fp32) -> xb (bf16), 8 elems/thread -----------------------------
__global__ __launch_bounds__(256) void k_cvt_x(const float* __restrict__ x,
                                               __bf16* __restrict__ xb) {
    long long i = ((long long)blockIdx.x * 256 + threadIdx.x) * 8;
    f32x4 a = *(const f32x4*)(x + i);
    f32x4 b = *(const f32x4*)(x + i + 4);
    bf16x8 o;
    o[0]=(__bf16)a[0]; o[1]=(__bf16)a[1]; o[2]=(__bf16)a[2]; o[3]=(__bf16)a[3];
    o[4]=(__bf16)b[0]; o[5]=(__bf16)b[1]; o[6]=(__bf16)b[2]; o[7]=(__bf16)b[3];
    *(bf16x8*)(xb + i) = o;
}

// --- K0b: pack f_w/g_w/h_w into one bf16 (640x512) matrix + fp32 bias ------
__global__ __launch_bounds__(256) void k_pack_w(const float* __restrict__ fw,
                                                const float* __restrict__ fb,
                                                const float* __restrict__ gw,
                                                const float* __restrict__ gb,
                                                const float* __restrict__ hw,
                                                const float* __restrict__ hb,
                                                __bf16* __restrict__ wb,
                                                float* __restrict__ bias) {
    int idx = blockIdx.x * 256 + threadIdx.x;
    if (idx < MCAT * CIN) {
        int o = idx >> 9, c = idx & 511;
        float v = (o < 64)  ? fw[o * CIN + c]
                : (o < 128) ? gw[(o - 64) * CIN + c]
                            : hw[(o - 128) * CIN + c];
        wb[idx] = (__bf16)v;
    }
    if (idx < MCAT) {
        bias[idx] = (idx < 64) ? fb[idx] : (idx < 128) ? gb[idx - 64] : hb[idx - 128];
    }
}

// --- K1: projection GEMM  y[b,o,s] = Wcat[o,:] . x[b,:,s] + bias[o] --------
// Block tile 64(M) x 256(N), 8 waves (2x4), wave tile 32x64, K-step 32.
__global__ __launch_bounds__(256) void k_proj(const __bf16* __restrict__ wb,
                                              const float* __restrict__ bias,
                                              const __bf16* __restrict__ xb,
                                              __bf16* __restrict__ y) {
    __shared__ __bf16 As[64 * LDK];
    __shared__ __bf16 Bs[256 * LDK];
    const int tid = threadIdx.x, lane = tid & 31, wid = tid >> 5;
    const int wm = wid & 1, wn = wid >> 1;
    const int bb = blockIdx.z;
    const int m0 = blockIdx.y * 64;
    const int s0 = blockIdx.x * 256;
    const __bf16* xbb = xb + (size_t)bb * CIN * WH;
    f32x8 acc[2][4];
    f32x8 zero = {0.f,0.f,0.f,0.f,0.f,0.f,0.f,0.f};
    #pragma unroll
    for (int i = 0; i < 2; i++)
        #pragma unroll
        for (int j = 0; j < 4; j++) acc[i][j] = zero;

    for (int k0 = 0; k0 < CIN; k0 += 32) {
        { // stage A (weights, row-major)
            int r = tid >> 2, kc = (tid & 3) * 8;
            bf16x8 v = *(const bf16x8*)(wb + (size_t)(m0 + r) * CIN + k0 + kc);
            *(bf16x8*)(&As[r * LDK + kc]) = v;
        }
        #pragma unroll
        for (int i = 0; i < 2; i++) { // stage B^T, two K-rows -> packed b32
            int idx = tid + i * 256;
            int kp = idx >> 5, nc = (idx & 31) * 8;   // kp: K-pair 0..15
            bf16x8 v0 = *(const bf16x8*)(xbb + (size_t)(k0 + 2*kp)     * WH + s0 + nc);
            bf16x8 v1 = *(const bf16x8*)(xbb + (size_t)(k0 + 2*kp + 1) * WH + s0 + nc);
            #pragma unroll
            for (int j = 0; j < 8; j++) {
                bf16x2 t; t[0] = v0[j]; t[1] = v1[j];
                *(bf16x2*)(&Bs[(nc + j) * LDK + 2*kp]) = t;
            }
        }
        __syncthreads();
        bf16x16 fa[2], fb2[4];
        #pragma unroll
        for (int mi = 0; mi < 2; mi++) fa[mi] = frag_a(As, wm * 32 + mi * 16, lane);
        #pragma unroll
        for (int ni = 0; ni < 4; ni++) fb2[ni] = frag_b(Bs, wn * 64 + ni * 16, lane);
        #pragma unroll
        for (int mi = 0; mi < 2; mi++)
            #pragma unroll
            for (int ni = 0; ni < 4; ni++)
                acc[mi][ni] = mma_bf16(fa[mi], fb2[ni], acc[mi][ni]);
        __syncthreads();
    }
    const int lr = lane & 15, hi = lane >> 4;
    #pragma unroll
    for (int mi = 0; mi < 2; mi++)
        #pragma unroll
        for (int ni = 0; ni < 4; ni++) {
            int col = s0 + wn * 64 + ni * 16 + lr;
            #pragma unroll
            for (int j = 0; j < 8; j++) {
                int row = m0 + wm * 32 + mi * 16 + hi * 8 + j;
                float v = acc[mi][ni][j] + bias[row];
                y[((size_t)bb * MCAT + row) * WH + col] = (__bf16)v;
            }
        }
}

// --- K2: logits GEMM  S[b,l,m] = sum_k f2[k,l] g2[k,m]  (M=N=1536, K=384) --
__global__ __launch_bounds__(256) void k_logits(const __bf16* __restrict__ y,
                                                float* __restrict__ S) {
    __shared__ __bf16 As[64 * LDK];
    __shared__ __bf16 Bs[256 * LDK];
    const int tid = threadIdx.x, lane = tid & 31, wid = tid >> 5;
    const int wm = wid & 1, wn = wid >> 1;
    const int bb = blockIdx.z;
    const int l0 = blockIdx.y * 64;    // M (l)
    const int m0 = blockIdx.x * 256;   // N (m)
    const __bf16* f2 = y + (size_t)bb * MCAT * WH;          // rows k -> +k*1536
    const __bf16* g2 = f2 + (size_t)64 * WH;
    f32x8 acc[2][4];
    f32x8 zero = {0.f,0.f,0.f,0.f,0.f,0.f,0.f,0.f};
    #pragma unroll
    for (int i = 0; i < 2; i++)
        #pragma unroll
        for (int j = 0; j < 4; j++) acc[i][j] = zero;

    for (int k0 = 0; k0 < KF; k0 += 32) {
        if (tid < 128) { // stage A^T: As[l][k] = f2[k, l], packed K-pairs
            int kp = tid >> 3, lc = (tid & 7) * 8;
            bf16x8 v0 = *(const bf16x8*)(f2 + (size_t)(k0 + 2*kp)     * LSEQ + l0 + lc);
            bf16x8 v1 = *(const bf16x8*)(f2 + (size_t)(k0 + 2*kp + 1) * LSEQ + l0 + lc);
            #pragma unroll
            for (int j = 0; j < 8; j++) {
                bf16x2 t; t[0] = v0[j]; t[1] = v1[j];
                *(bf16x2*)(&As[(lc + j) * LDK + 2*kp]) = t;
            }
        }
        #pragma unroll
        for (int i = 0; i < 2; i++) { // stage B^T: Bs[m][k] = g2[k, m]
            int idx = tid + i * 256;
            int kp = idx >> 5, nc = (idx & 31) * 8;
            bf16x8 v0 = *(const bf16x8*)(g2 + (size_t)(k0 + 2*kp)     * LSEQ + m0 + nc);
            bf16x8 v1 = *(const bf16x8*)(g2 + (size_t)(k0 + 2*kp + 1) * LSEQ + m0 + nc);
            #pragma unroll
            for (int j = 0; j < 8; j++) {
                bf16x2 t; t[0] = v0[j]; t[1] = v1[j];
                *(bf16x2*)(&Bs[(nc + j) * LDK + 2*kp]) = t;
            }
        }
        __syncthreads();
        bf16x16 fa[2], fb2[4];
        #pragma unroll
        for (int mi = 0; mi < 2; mi++) fa[mi] = frag_a(As, wm * 32 + mi * 16, lane);
        #pragma unroll
        for (int ni = 0; ni < 4; ni++) fb2[ni] = frag_b(Bs, wn * 64 + ni * 16, lane);
        #pragma unroll
        for (int mi = 0; mi < 2; mi++)
            #pragma unroll
            for (int ni = 0; ni < 4; ni++)
                acc[mi][ni] = mma_bf16(fa[mi], fb2[ni], acc[mi][ni]);
        __syncthreads();
    }
    const int lr = lane & 15, hi = lane >> 4;
    #pragma unroll
    for (int mi = 0; mi < 2; mi++)
        #pragma unroll
        for (int ni = 0; ni < 4; ni++) {
            int col = m0 + wn * 64 + ni * 16 + lr;
            #pragma unroll
            for (int j = 0; j < 8; j++) {
                int row = l0 + wm * 32 + mi * 16 + hi * 8 + j;
                S[((size_t)bb * LSEQ + row) * LSEQ + col] = acc[mi][ni][j];
            }
        }
}

// --- K3: row softmax over 1536 (one wave32 per row), bf16 output -----------
__global__ __launch_bounds__(256) void k_softmax(const float* __restrict__ S,
                                                 __bf16* __restrict__ P) {
    const int lane = threadIdx.x & 31, wid = threadIdx.x >> 5;
    const long long row = (long long)blockIdx.x * 8 + wid;  // [0, 16*1536)
    const float* sr = S + row * LSEQ;
    float v[48];
    float mx = -3.402823466e38f;
    #pragma unroll
    for (int i = 0; i < 48; i++) { v[i] = sr[lane + i * 32]; mx = fmaxf(mx, v[i]); }
    #pragma unroll
    for (int o = 16; o > 0; o >>= 1) mx = fmaxf(mx, __shfl_xor(mx, o, 32));
    float sum = 0.f;
    #pragma unroll
    for (int i = 0; i < 48; i++) { v[i] = __expf(v[i] - mx); sum += v[i]; }
    #pragma unroll
    for (int o = 16; o > 0; o >>= 1) sum += __shfl_xor(sum, o, 32);
    const float inv = 1.0f / sum;
    __bf16* pr = P + row * LSEQ;
    #pragma unroll
    for (int i = 0; i < 48; i++) pr[lane + i * 32] = (__bf16)(v[i] * inv);
}

// --- K4: out GEMM  sa[b,c2,m] = sum_l h2[c2,l] P[b,l,m]; out = g*sa + x ----
// M=3072, N=1536, K=1536. Block tile 128x128, waves 4x2, wave tile 32x64.
// A tiles (h2, 128 rows x 32 K, row stride 1536) are DMA'd by the TDM into a
// double-buffered padded LDS region: wave 0 issues tensor_load_to_lds for
// tile k+1 before computing tile k and waits TENSORcnt<=1 (in-order TDM
// completion), overlapping the DMA with 8 WMMAs + B staging.
__global__ __launch_bounds__(256) void k_out(const __bf16* __restrict__ y,
                                             const __bf16* __restrict__ P,
                                             const float* __restrict__ x,
                                             const float* __restrict__ gamma,
                                             float* __restrict__ out) {
    __shared__ __bf16 As[2][128 * LDK];
    __shared__ __bf16 Bs[128 * LDK];
    const int tid = threadIdx.x, lane = tid & 31, wid = tid >> 5;
    const int wm = wid & 3, wn = wid >> 2;
    const int bb = blockIdx.z;
    const int m0 = blockIdx.y * 128;   // c2
    const int n0 = blockIdx.x * 128;   // m
    const __bf16* h2 = y + ((size_t)bb * MCAT + 128) * WH;  // rows c2 -> +c2*1536
    const __bf16* Pb = P + (size_t)bb * LSEQ * LSEQ;
    const unsigned long long ha =
        (unsigned long long)(size_t)(h2 + (size_t)m0 * LSEQ);
    const unsigned as_lds0 = (unsigned)(size_t)(&As[0][0]);
    const unsigned as_lds1 = (unsigned)(size_t)(&As[1][0]);
    f32x8 acc[2][4];
    f32x8 zero = {0.f,0.f,0.f,0.f,0.f,0.f,0.f,0.f};
    #pragma unroll
    for (int i = 0; i < 2; i++)
        #pragma unroll
        for (int j = 0; j < 4; j++) acc[i][j] = zero;

    if (wid == 0) tdm_load_tile_a(as_lds0, ha * 1u + 0);  // prologue: tile 0

    for (int k0 = 0; k0 < LSEQ; k0 += 32) {
        const int cur = (k0 >> 5) & 1;
        if (k0 + 32 < LSEQ) {  // speculative L2 prefetch of next B tile
            __builtin_prefetch(Pb + (size_t)(k0 + 32 + (tid & 31)) * LSEQ + n0, 0, 1);
        }
        { // stage B^T: Bs[m][l], packed K-pairs (one task per thread)
            int kp = tid >> 4, nc = (tid & 15) * 8;   // kp 0..15, nc 0..120
            bf16x8 v0 = *(const bf16x8*)(Pb + (size_t)(k0 + 2*kp)     * LSEQ + n0 + nc);
            bf16x8 v1 = *(const bf16x8*)(Pb + (size_t)(k0 + 2*kp + 1) * LSEQ + n0 + nc);
            #pragma unroll
            for (int j = 0; j < 8; j++) {
                bf16x2 t; t[0] = v0[j]; t[1] = v1[j];
                *(bf16x2*)(&Bs[(nc + j) * LDK + 2*kp]) = t;
            }
        }
        if (wid == 0) {
            if (k0 + 32 < LSEQ) { // issue next A tile into the other buffer
                tdm_load_tile_a(cur ? as_lds0 : as_lds1,
                                ha + (unsigned long long)(k0 + 32) * 2u);
                __builtin_amdgcn_s_wait_tensorcnt(1); // current tile complete
            } else {
                __builtin_amdgcn_s_wait_tensorcnt(0); // last tile complete
            }
        }
        __syncthreads();
        const __bf16* At = &As[cur][0];
        bf16x16 fa[2], fb2[4];
        #pragma unroll
        for (int mi = 0; mi < 2; mi++) fa[mi] = frag_a(At, wm * 32 + mi * 16, lane);
        #pragma unroll
        for (int ni = 0; ni < 4; ni++) fb2[ni] = frag_b(Bs, wn * 64 + ni * 16, lane);
        #pragma unroll
        for (int mi = 0; mi < 2; mi++)
            #pragma unroll
            for (int ni = 0; ni < 4; ni++)
                acc[mi][ni] = mma_bf16(fa[mi], fb2[ni], acc[mi][ni]);
        __syncthreads();
    }
    const float g0 = gamma[0];
    const int lr = lane & 15, hi = lane >> 4;
    #pragma unroll
    for (int mi = 0; mi < 2; mi++)
        #pragma unroll
        for (int ni = 0; ni < 4; ni++) {
            int col = n0 + wn * 64 + ni * 16 + lr;
            #pragma unroll
            for (int j = 0; j < 8; j++) {
                int row = m0 + wm * 32 + mi * 16 + hi * 8 + j;
                size_t o = ((size_t)bb * KH + row) * (size_t)LSEQ + col;
                out[o] = g0 * acc[mi][ni][j] + x[o];
            }
        }
}

// ---------------------------------------------------------------------------
extern "C" void kernel_launch(void* const* d_in, const int* in_sizes, int n_in,
                              void* d_out, int out_size, void* d_ws, size_t ws_size,
                              hipStream_t stream) {
    (void)in_sizes; (void)n_in; (void)out_size; (void)ws_size;
    const float* x     = (const float*)d_in[0];
    const float* f_w   = (const float*)d_in[1];
    const float* f_b   = (const float*)d_in[2];
    const float* g_w   = (const float*)d_in[3];
    const float* g_b   = (const float*)d_in[4];
    const float* h_w   = (const float*)d_in[5];
    const float* h_b   = (const float*)d_in[6];
    const float* gamma = (const float*)d_in[7];
    float* out = (float*)d_out;

    char* ws = (char*)d_ws;
    size_t off = 0;
    __bf16* xb = (__bf16*)(ws + off); off += (size_t)NB * CIN * WH * 2;   // 151.0 MB
    __bf16* wb = (__bf16*)(ws + off); off += (size_t)MCAT * CIN * 2;      // 0.66 MB
    float*  bias = (float*)(ws + off); off += 4096;
    __bf16* y  = (__bf16*)(ws + off); off += (size_t)NB * MCAT * WH * 2;  // 188.7 MB
    float*  S  = (float*)(ws + off);  off += (size_t)NB * LSEQ * LSEQ * 4; // 151.0 MB
    __bf16* P  = (__bf16*)(ws + off); off += (size_t)NB * LSEQ * LSEQ * 2; // 75.5 MB

    k_cvt_x  <<<dim3((NB * CIN * WH) / (256 * 8)), dim3(256), 0, stream>>>(x, xb);
    k_pack_w <<<dim3((MCAT * CIN + 255) / 256), dim3(256), 0, stream>>>(
        f_w, f_b, g_w, g_b, h_w, h_b, wb, bias);
    k_proj   <<<dim3(WH / 256, MCAT / 64, NB), dim3(256), 0, stream>>>(wb, bias, xb, y);
    k_logits <<<dim3(LSEQ / 256, LSEQ / 64, NB), dim3(256), 0, stream>>>(y, S);
    k_softmax<<<dim3(NB * LSEQ / 8), dim3(256), 0, stream>>>(S, P);
    k_out    <<<dim3(LSEQ / 128, KH / 128, NB), dim3(256), 0, stream>>>(y, P, x, gamma, out);
}
